// InvariantPointAttention_1305670058169
// MI455X (gfx1250) — matrix operand order repaired and checked by reading, not compiled, not run on hardware
//
#include <hip/hip_runtime.h>
#include <hip/hip_bf16.h>
#include <math.h>
#include <stdint.h>

// ---------------- problem constants ----------------
constexpr int kN  = 512;
constexpr int kCZ = 128;
constexpr int kH  = 12;
constexpr int kC  = 16;
constexpr int kPQ = 4;
constexpr int kPV = 8;
constexpr int kNN = kN * kN;                      // 262144
constexpr int kCAT = kH * (kC + kPV * 4 + kCZ);   // 2112
constexpr float kINF = 100000.0f;
constexpr float kEPS = 1e-8f;
constexpr float kSQK = 0.14433756729740643f;      // sqrt(1/48)
constexpr float kSB  = 0.5773502691896258f;       // sqrt(1/3)
constexpr float kSPT = 0.13608276348795434f;      // sqrt(1/54)

// ---------------- workspace layout (floats) ----------------
constexpr size_t OFF_Q    = 0;                          // 512*192
constexpr size_t OFF_KV   = OFF_Q    + 512 * 192;       // 512*384
constexpr size_t OFF_QPL  = OFF_KV   + 512 * 384;       // 512*144
constexpr size_t OFF_KVPL = OFF_QPL  + 512 * 144;       // 512*432
constexpr size_t OFF_QPTS = OFF_KVPL + 512 * 432;       // 512*144
constexpr size_t OFF_KPTS = OFF_QPTS + 512 * 144;       // 512*144
constexpr size_t OFF_VPTS = OFF_KPTS + 512 * 144;       // 512*288
constexpr size_t OFF_QN   = OFF_VPTS + 512 * 288;       // 512*12
constexpr size_t OFF_KN   = OFF_QN   + 512 * 12;        // 512*12
constexpr size_t OFF_WBP  = OFF_KN   + 512 * 12;        // 128*16 padded W_b
constexpr size_t OFF_BBP  = OFF_WBP  + 128 * 16;        // 64 (16 used)
constexpr size_t OFF_BIAS = OFF_BBP  + 64;              // 16*N*N (12 used)
constexpr size_t OFF_A    = OFF_BIAS + (size_t)16 * kNN; // 12*N*N
constexpr size_t OFF_CAT  = OFF_A    + (size_t)kH * kNN; // 512*2112
// total ~9.32M floats ~ 37.3 MB

// ---------------- WMMA helpers ----------------
typedef __attribute__((ext_vector_type(16))) _Float16 v16h;
typedef __attribute__((ext_vector_type(8)))  float    v8f;

__device__ __forceinline__ v8f wmma16(v16h a, v16h b, v8f c) {
  // D = A(16x32) * B(32x16) + C, fp16 in / fp32 accumulate
  return __builtin_amdgcn_wmma_f32_16x16x32_f16(false, a, false, b, (short)0, c,
                                                false, false);
}
// A-fragment k index for element e (ISA 16-bit A 16x32 layout)
__device__ __forceinline__ int a_k(int lane, int e) {
  return (e >> 3) * 16 + (lane >> 4) * 8 + (e & 7);
}
// B-fragment k index for element e (same k order as a_k)
__device__ __forceinline__ int b_k(int lane, int e) {
  return (e >> 3) * 16 + (lane >> 4) * 8 + (e & 7);
}

// gfx1250 async copy: 16 bytes global -> LDS, tracked by ASYNCcnt.
__device__ __forceinline__ void async_copy16(const float* gsrc, float* ldst) {
  unsigned lds32 = (unsigned)(uintptr_t)ldst;          // low 32 bits = DS offset
  unsigned long long g64 = (unsigned long long)(uintptr_t)gsrc;
  asm volatile("global_load_async_to_lds_b128 %0, %1, off"
               :: "v"(lds32), "v"(g64) : "memory");
}
__device__ __forceinline__ void wait_async0() {
  asm volatile("s_wait_asynccnt 0x0" ::: "memory");
}

// =====================================================================
// Generic one-wave 16x16-tile GEMM: out = alpha*(A@B + bias)
// Preconditions (guaranteed by caller): M%16==0, N%16==0, K%32==0,
// lda/ldb multiples of 4 -> all float4 accesses 16B aligned.
// =====================================================================
__global__ void gemm_wmma_kernel(const float* __restrict__ A,
                                 const float* __restrict__ B,
                                 const float* __restrict__ bias,
                                 float* __restrict__ out,
                                 int M, int N, int K,
                                 int lda, int ldb, int ldo,
                                 float alpha, int transpose_out) {
  __shared__ float As[16][32];
  __shared__ float Bs[32][16];
  const int t    = threadIdx.x;          // 0..31, one wave
  const int m0   = blockIdx.y * 16;
  const int n0   = blockIdx.x * 16;
  const int nn   = t & 15;
  const int half = t >> 4;

  v8f acc = {};
  for (int k0 = 0; k0 < K; k0 += 32) {
    // A tile 16x32 = 128 float4; thread t: rows (t>>3)+4i, float4-col t&7
#pragma unroll
    for (int i = 0; i < 4; ++i) {
      int r = (t >> 3) + i * 4;
      int cv = t & 7;
      *(float4*)&As[r][cv * 4] =
          *(const float4*)&A[(size_t)(m0 + r) * lda + k0 + cv * 4];
    }
    // B tile 32x16 = 128 float4; thread t: row t, 4 float4
#pragma unroll
    for (int i = 0; i < 4; ++i)
      *(float4*)&Bs[t][i * 4] =
          *(const float4*)&B[(size_t)(k0 + t) * ldb + n0 + i * 4];
    if (k0 + 32 < K)   // prefetch next A K-tile -> global_prefetch_b8
      __builtin_prefetch(&A[(size_t)(m0 + (t >> 3)) * lda + k0 + 32 + (t & 7) * 4], 0, 1);
    __syncthreads();
    v16h af, bf;
#pragma unroll
    for (int e = 0; e < 16; ++e) {
      af[e] = (_Float16)As[nn][a_k(t, e)];
      bf[e] = (_Float16)Bs[b_k(t, e)][nn];
    }
    acc = wmma16(af, bf, acc);
    __syncthreads();
  }
#pragma unroll
  for (int r = 0; r < 8; ++r) {
    int m = r + half * 8;
    int gm = m0 + m, gn = n0 + nn;
    float v = acc[r];
    if (bias) v += bias[gn];
    v *= alpha;
    if (transpose_out) out[(size_t)gn * ldo + gm] = v;
    else               out[(size_t)gm * ldo + gn] = v;
  }
}

// =====================================================================
// Pad W_b (128x12) -> (128x16), b_b (12) -> (16) with zeros.
// =====================================================================
__global__ void pad_wb_kernel(const float* __restrict__ Wb,
                              const float* __restrict__ bb,
                              float* __restrict__ Wbp,
                              float* __restrict__ bbp) {
  int t = blockIdx.x * blockDim.x + threadIdx.x;
  if (t < 128 * 16) {
    int r = t >> 4, c = t & 15;
    Wbp[t] = (c < kH) ? Wb[r * kH + c] : 0.0f;
  }
  if (t < 16) bbp[t] = (t < kH) ? bb[t] : 0.0f;
}

// =====================================================================
// Rotate projected points into the global frame + point norms.
// =====================================================================
__global__ void points_kernel(const float* __restrict__ rot,
                              const float* __restrict__ trans,
                              const float* __restrict__ qpl,   // (N,144)
                              const float* __restrict__ kvpl,  // (N,432)
                              float* __restrict__ qpts,        // (N,H,PQ,3)
                              float* __restrict__ kpts,        // (N,H,PQ,3)
                              float* __restrict__ vpts,        // (N,H,PV,3)
                              float* __restrict__ qn,          // (N,H)
                              float* __restrict__ kn) {        // (N,H)
  int idx = blockIdx.x * blockDim.x + threadIdx.x;
  if (idx >= kN * kH) return;
  int n = idx / kH, h = idx % kH;
  float R[9], T[3];
#pragma unroll
  for (int i = 0; i < 9; ++i) R[i] = rot[n * 9 + i];
#pragma unroll
  for (int i = 0; i < 3; ++i) T[i] = trans[n * 3 + i];

  float qnorm = 0.0f;
#pragma unroll
  for (int p = 0; p < kPQ; ++p) {
    int hp = h * kPQ + p;
    float x = qpl[n * 144 + 0 * 48 + hp];
    float y = qpl[n * 144 + 1 * 48 + hp];
    float zc = qpl[n * 144 + 2 * 48 + hp];
    float gx = R[0] * x + R[1] * y + R[2] * zc + T[0];
    float gy = R[3] * x + R[4] * y + R[5] * zc + T[1];
    float gz = R[6] * x + R[7] * y + R[8] * zc + T[2];
    size_t base = ((size_t)n * kH + h) * 12 + p * 3;
    qpts[base + 0] = gx; qpts[base + 1] = gy; qpts[base + 2] = gz;
    qnorm += gx * gx + gy * gy + gz * gz;
  }
  qn[n * kH + h] = qnorm;

  float knorm = 0.0f;
#pragma unroll
  for (int pt = 0; pt < kPQ + kPV; ++pt) {
    int hp = h * (kPQ + kPV) + pt;
    float x = kvpl[n * 432 + 0 * 144 + hp];
    float y = kvpl[n * 432 + 1 * 144 + hp];
    float zc = kvpl[n * 432 + 2 * 144 + hp];
    float gx = R[0] * x + R[1] * y + R[2] * zc + T[0];
    float gy = R[3] * x + R[4] * y + R[5] * zc + T[1];
    float gz = R[6] * x + R[7] * y + R[8] * zc + T[2];
    if (pt < kPQ) {
      size_t base = ((size_t)n * kH + h) * 12 + pt * 3;
      kpts[base + 0] = gx; kpts[base + 1] = gy; kpts[base + 2] = gz;
      knorm += gx * gx + gy * gy + gz * gz;
    } else {
      size_t base = ((size_t)n * kH + h) * 24 + (pt - kPQ) * 3;
      vpts[base + 0] = gx; vpts[base + 1] = gy; vpts[base + 2] = gz;
    }
  }
  kn[n * kH + h] = knorm;
}

// =====================================================================
// Fused attention per (i-tile of 16 rows, head h).
// Combined logit WMMA: A-row = [q*sqk | q_pts*hw | 0],
//                      B-col = [k     | k_pts    | 0].
// + bias_t - 0.5*hw*(|q_pts|^2+|k_pts|^2) + mask; softmax over 512;
// o = a@v, o_pt = a@v_pts (WMMA); o_pt rotated back to local frame.
// =====================================================================
__global__ void attn_kernel(const float* __restrict__ q,      // (N,H,C)
                            const float* __restrict__ kv,     // (N,H,2C)
                            const float* __restrict__ qpts,
                            const float* __restrict__ kpts,
                            const float* __restrict__ vpts,
                            const float* __restrict__ qn,
                            const float* __restrict__ kn,
                            const float* __restrict__ bias_t, // (16,N,N)
                            const float* __restrict__ mask,   // (N)
                            const float* __restrict__ hwts,   // (H)
                            const float* __restrict__ rot,
                            const float* __restrict__ trans,
                            float* __restrict__ a_out,        // (H,N,N)
                            float* __restrict__ cat) {        // (N,2112)
  __shared__ float Atile[16][32];
  __shared__ float BtT[16][32];    // logits B, layout [j-local][channel]
  __shared__ float Btile[32][16];  // o/o_pt B, layout [j-local][col]
  __shared__ float L[16][512];
  __shared__ float Sopt[16][32];
  __shared__ float red[2][16];

  const int it = blockIdx.x;            // 0..31
  const int h  = blockIdx.y;            // 0..11
  const int t  = threadIdx.x;           // 0..31 (one wave)
  const int i0 = it * 16;
  const int nn = t & 15;
  const int half = t >> 4;
  const float4 zero4{0.0f, 0.0f, 0.0f, 0.0f};

  const float hw = log1pf(expf(hwts[h])) * kSPT;   // softplus * sqrt(1/54)

  // -------- A tile: [q*sqk (16) | q_pts*hw (12) | 0 (4)] --------
  {
    int r = nn;
    if (half == 0) {
      const float4* qp = (const float4*)&q[(size_t)(i0 + r) * 192 + h * 16];
#pragma unroll
      for (int i = 0; i < 4; ++i) {
        float4 v = qp[i];
        v.x *= kSQK; v.y *= kSQK; v.z *= kSQK; v.w *= kSQK;
        *(float4*)&Atile[r][i * 4] = v;
      }
    } else {
      const float4* pp = (const float4*)&qpts[((size_t)(i0 + r) * kH + h) * 12];
#pragma unroll
      for (int i = 0; i < 3; ++i) {
        float4 v = pp[i];
        v.x *= hw; v.y *= hw; v.z *= hw; v.w *= hw;
        *(float4*)&Atile[r][16 + i * 4] = v;
      }
      *(float4*)&Atile[r][28] = zero4;
    }
  }
  __syncthreads();
  v16h afrag;
#pragma unroll
  for (int e = 0; e < 16; ++e) afrag[e] = (_Float16)Atile[nn][a_k(t, e)];

  // -------- logits over all j tiles --------
  for (int jt = 0; jt < 32; ++jt) {
    const int j0 = jt * 16;
    {
      int c = nn, j = j0 + c;
      if (half == 0) {
        const float4* kp = (const float4*)&kv[(size_t)j * 384 + h * 32];
#pragma unroll
        for (int i = 0; i < 4; ++i) *(float4*)&BtT[c][i * 4] = kp[i];
      } else {
        const float4* pp = (const float4*)&kpts[((size_t)j * kH + h) * 12];
#pragma unroll
        for (int i = 0; i < 3; ++i) *(float4*)&BtT[c][16 + i * 4] = pp[i];
        *(float4*)&BtT[c][28] = zero4;
      }
    }
    __syncthreads();
    v16h bfrag;
#pragma unroll
    for (int e = 0; e < 16; ++e) bfrag[e] = (_Float16)BtT[nn][b_k(t, e)];
    v8f c0 = {};
    c0 = wmma16(afrag, bfrag, c0);
#pragma unroll
    for (int r = 0; r < 8; ++r) {
      int m = r + half * 8;
      int gi = i0 + m, gj = j0 + nn;
      float v = c0[r];
      v += bias_t[(size_t)h * kNN + (size_t)gi * kN + gj];
      v += -0.5f * hw * (qn[gi * kH + h] + kn[gj * kH + h]);
      v += kINF * (mask[gi] * mask[gj] - 1.0f);
      L[m][gj] = v;
    }
    __syncthreads();
  }

  // -------- softmax over 512 columns (2 threads per row) --------
  {
    int row = nn, seg = half;
    float mx = -3.0e38f;
    for (int c = 0; c < 256; ++c) mx = fmaxf(mx, L[row][seg * 256 + c]);
    red[seg][row] = mx;
    __syncthreads();
    mx = fmaxf(red[0][row], red[1][row]);
    __syncthreads();
    float sum = 0.0f;
    for (int c = 0; c < 256; ++c) {
      float e = __expf(L[row][seg * 256 + c] - mx);
      L[row][seg * 256 + c] = e;
      sum += e;
    }
    red[seg][row] = sum;
    __syncthreads();
    sum = red[0][row] + red[1][row];
    float inv = 1.0f / sum;
    for (int c = 0; c < 256; ++c) {
      float p = L[row][seg * 256 + c] * inv;
      L[row][seg * 256 + c] = p;
      a_out[(size_t)h * kNN + (size_t)(i0 + row) * kN + seg * 256 + c] = p;
    }
    __syncthreads();
  }

  // -------- o = a@v, o_pt = a@v_pts (16 K-chunks of 32 j) --------
  v8f aco = {}, acp0 = {}, acp1 = {};
  for (int kt = 0; kt < 16; ++kt) {
    v16h pfrag;
#pragma unroll
    for (int e = 0; e < 16; ++e)
      pfrag[e] = (_Float16)L[nn][kt * 32 + a_k(t, e)];
    const int jrow = kt * 32 + t;       // this thread's B row (j index)

    // v tile
    {
      const float4* vp = (const float4*)&kv[(size_t)jrow * 384 + h * 32 + 16];
#pragma unroll
      for (int i = 0; i < 4; ++i) *(float4*)&Btile[t][i * 4] = vp[i];
    }
    __syncthreads();
    {
      v16h bf;
#pragma unroll
      for (int e = 0; e < 16; ++e) bf[e] = (_Float16)Btile[b_k(t, e)][nn];
      aco = wmma16(pfrag, bf, aco);
    }
    __syncthreads();

    // v_pts cols 0..15
    {
      const float4* pp = (const float4*)&vpts[((size_t)jrow * kH + h) * 24];
#pragma unroll
      for (int i = 0; i < 4; ++i) *(float4*)&Btile[t][i * 4] = pp[i];
    }
    __syncthreads();
    {
      v16h bf;
#pragma unroll
      for (int e = 0; e < 16; ++e) bf[e] = (_Float16)Btile[b_k(t, e)][nn];
      acp0 = wmma16(pfrag, bf, acp0);
    }
    __syncthreads();

    // v_pts cols 16..23 (pad to 16)
    {
      const float4* pp = (const float4*)&vpts[((size_t)jrow * kH + h) * 24 + 16];
      *(float4*)&Btile[t][0] = pp[0];
      *(float4*)&Btile[t][4] = pp[1];
      *(float4*)&Btile[t][8] = zero4;
      *(float4*)&Btile[t][12] = zero4;
    }
    __syncthreads();
    {
      v16h bf;
#pragma unroll
      for (int e = 0; e < 16; ++e) bf[e] = (_Float16)Btile[b_k(t, e)][nn];
      acp1 = wmma16(pfrag, bf, acp1);
    }
    __syncthreads();
  }

  // -------- epilogue: o + stage o_pt --------
#pragma unroll
  for (int r = 0; r < 8; ++r) {
    int m = r + half * 8;
    cat[(size_t)(i0 + m) * kCAT + h * kC + nn] = aco[r];
    Sopt[m][nn] = acp0[r];
    Sopt[m][16 + nn] = acp1[r];   // cols 24..31 scratch, never read
  }
  __syncthreads();

  // -------- epilogue: o_pt -> local frame + norms --------
  for (int task = t; task < 16 * kPV; task += 32) {
    int row = task >> 3, p = task & 7;
    int gi = i0 + row;
    float gx = Sopt[row][p * 3 + 0];
    float gy = Sopt[row][p * 3 + 1];
    float gz = Sopt[row][p * 3 + 2];
    const float* R = rot + gi * 9;
    float dx = gx - trans[gi * 3 + 0];
    float dy = gy - trans[gi * 3 + 1];
    float dz = gz - trans[gi * 3 + 2];
    float lx = R[0] * dx + R[3] * dy + R[6] * dz;   // rot^T
    float ly = R[1] * dx + R[4] * dy + R[7] * dz;
    float lz = R[2] * dx + R[5] * dy + R[8] * dz;
    size_t base = (size_t)gi * kCAT;
    int hp = h * kPV + p;
    cat[base + 192 + 0 * 96 + hp] = lx;
    cat[base + 192 + 1 * 96 + hp] = ly;
    cat[base + 192 + 2 * 96 + hp] = lz;
    cat[base + 480 + hp] = sqrtf(lx * lx + ly * ly + lz * lz + kEPS);
  }
}

// =====================================================================
// o_pair[i] = a[:,i,:] @ z[i]  (one block per residue i)
// M=16 (12 heads + pad), K=512 (j), N=128 (z channels).
// z staged to LDS with gfx1250 global_load_async_to_lds_b128.
// =====================================================================
__global__ void opair_kernel(const float* __restrict__ a_ws,   // (H,N,N)
                             const float* __restrict__ z,      // (N,N,CZ)
                             float* __restrict__ cat) {
  __shared__ float Az[16][32];
  __shared__ float Bz[32][128];
  const int i = blockIdx.x;
  const int t = threadIdx.x;
  const int nn = t & 15;
  const int half = t >> 4;

  v8f acc[8];
#pragma unroll
  for (int nc = 0; nc < 8; ++nc) acc[nc] = (v8f){};

  for (int kt = 0; kt < 16; ++kt) {
    const int j0 = kt * 32;
    // async-stage z rows (32 x 128 floats): thread t -> row t, 32 x b128
    const size_t zrow = ((size_t)i * kN + j0 + t) * kCZ;
#pragma unroll
    for (int c = 0; c < 32; ++c)
      async_copy16(&z[zrow + c * 4], &Bz[t][c * 4]);
    // stage attention rows (16 x 32): thread t -> row t&15, cols (t>>4)*16..
    {
      int r = nn;
#pragma unroll
      for (int v = 0; v < 4; ++v) {
        int col = half * 16 + v * 4;
        float4 val = (r < kH)
            ? *(const float4*)&a_ws[(size_t)r * kNN + (size_t)i * kN + j0 + col]
            : float4{0.0f, 0.0f, 0.0f, 0.0f};
        *(float4*)&Az[r][col] = val;
      }
    }
    wait_async0();
    __syncthreads();

    v16h af;
#pragma unroll
    for (int e = 0; e < 16; ++e) af[e] = (_Float16)Az[nn][a_k(t, e)];
#pragma unroll
    for (int nc = 0; nc < 8; ++nc) {
      v16h bf;
#pragma unroll
      for (int e = 0; e < 16; ++e)
        bf[e] = (_Float16)Bz[b_k(t, e)][nc * 16 + nn];
      acc[nc] = wmma16(af, bf, acc[nc]);
    }
    __syncthreads();
  }

#pragma unroll
  for (int nc = 0; nc < 8; ++nc) {
#pragma unroll
    for (int r = 0; r < 8; ++r) {
      int hh = r + half * 8;
      if (hh < kH)
        cat[(size_t)i * kCAT + 576 + hh * kCZ + nc * 16 + nn] = acc[nc][r];
    }
  }
}

// =====================================================================
extern "C" void kernel_launch(void* const* d_in, const int* in_sizes, int n_in,
                              void* d_out, int out_size, void* d_ws, size_t ws_size,
                              hipStream_t stream) {
  (void)in_sizes; (void)n_in; (void)out_size; (void)ws_size;
  const float* s     = (const float*)d_in[0];
  const float* z     = (const float*)d_in[1];
  const float* rot   = (const float*)d_in[2];
  const float* trans = (const float*)d_in[3];
  const float* mask  = (const float*)d_in[4];
  const float* W_q   = (const float*)d_in[5];
  const float* b_q   = (const float*)d_in[6];
  const float* W_kv  = (const float*)d_in[7];
  const float* b_kv  = (const float*)d_in[8];
  const float* W_qp  = (const float*)d_in[9];
  const float* b_qp  = (const float*)d_in[10];
  const float* W_kvp = (const float*)d_in[11];
  const float* b_kvp = (const float*)d_in[12];
  const float* W_b   = (const float*)d_in[13];
  const float* b_b   = (const float*)d_in[14];
  const float* hwts  = (const float*)d_in[15];
  const float* W_out = (const float*)d_in[16];
  const float* b_out = (const float*)d_in[17];
  float* ws  = (float*)d_ws;
  float* out = (float*)d_out;

  float* ws_q    = ws + OFF_Q;
  float* ws_kv   = ws + OFF_KV;
  float* ws_qpl  = ws + OFF_QPL;
  float* ws_kvpl = ws + OFF_KVPL;
  float* ws_qpts = ws + OFF_QPTS;
  float* ws_kpts = ws + OFF_KPTS;
  float* ws_vpts = ws + OFF_VPTS;
  float* ws_qn   = ws + OFF_QN;
  float* ws_kn   = ws + OFF_KN;
  float* ws_wbp  = ws + OFF_WBP;
  float* ws_bbp  = ws + OFF_BBP;
  float* ws_bias = ws + OFF_BIAS;
  float* ws_a    = ws + OFF_A;
  float* ws_cat  = ws + OFF_CAT;

  // 0) pad W_b/b_b to 16 columns (removes all GEMM bounds checks)
  pad_wb_kernel<<<8, 256, 0, stream>>>(W_b, b_b, ws_wbp, ws_bbp);

  // 1) projections: s @ {W_q, W_kv, W_qp, W_kvp} + biases
  gemm_wmma_kernel<<<dim3(192 / 16, 32), 32, 0, stream>>>(
      s, W_q, b_q, ws_q, 512, 192, 384, 384, 192, 192, 1.0f, 0);
  gemm_wmma_kernel<<<dim3(384 / 16, 32), 32, 0, stream>>>(
      s, W_kv, b_kv, ws_kv, 512, 384, 384, 384, 384, 384, 1.0f, 0);
  gemm_wmma_kernel<<<dim3(144 / 16, 32), 32, 0, stream>>>(
      s, W_qp, b_qp, ws_qpl, 512, 144, 384, 384, 144, 144, 1.0f, 0);
  gemm_wmma_kernel<<<dim3(432 / 16, 32), 32, 0, stream>>>(
      s, W_kvp, b_kvp, ws_kvpl, 512, 432, 384, 384, 432, 432, 1.0f, 0);

  // 2) global-frame points + norms
  points_kernel<<<(kN * kH + 63) / 64, 64, 0, stream>>>(
      rot, trans, ws_qpl, ws_kvpl, ws_qpts, ws_kpts, ws_vpts, ws_qn, ws_kn);

  // 3) bias_t[h][i][j] = sqrt(1/3) * (z @ W_b + b_b)  (transposed store,
  //    padded to 16 head-columns; cols 12..15 land in scratch rows)
  gemm_wmma_kernel<<<dim3(1, kNN / 16), 32, 0, stream>>>(
      z, ws_wbp, ws_bbp, ws_bias, kNN, 16, kCZ, kCZ, 16, kNN, kSB, 1);

  // 4) fused attention per (i-tile, head)
  attn_kernel<<<dim3(32, kH), 32, 0, stream>>>(
      ws_q, ws_kv, ws_qpts, ws_kpts, ws_vpts, ws_qn, ws_kn,
      ws_bias, mask, hwts, rot, trans, ws_a, ws_cat);

  // 5) o_pair (second z pass, async-to-LDS staging)
  opair_kernel<<<kN, 32, 0, stream>>>(ws_a, z, ws_cat);

  // 6) final projection: cat @ W_out + b_out -> d_out
  gemm_wmma_kernel<<<dim3(384 / 16, 32), 32, 0, stream>>>(
      ws_cat, W_out, b_out, out, 512, 384, kCAT, kCAT, 384, 384, 1.0f, 0);
}
